// VectorModalityEncoder_62148176773395
// MI455X (gfx1250) — matrix operand order repaired and checked by reading, not compiled, not run on hardware
//
#include <hip/hip_runtime.h>
#include <hip/hip_bf16.h>
#include <stdint.h>

// ---------------- problem constants ----------------
#define BB   32
#define TT   1024
#define DD   256
#define NHH  8
#define NLL  4
#define SDD  11
#define DFFF 1024
#define LVIS 512
#define DHH  32

// ---------------- WMMA fragment helpers ----------------
typedef __attribute__((ext_vector_type(16))) _Float16 v16h;
typedef __attribute__((ext_vector_type(8)))  float    v8f;
typedef __attribute__((ext_vector_type(4)))  unsigned uint4v;
typedef __attribute__((ext_vector_type(2)))  unsigned uint2v;

union FragU { v16h h; uint4v u4[2]; };

#define TROW 16   // dwords per LDS tile row (32 halves = 64B, b128-aligned)

// A-matrix 16x32 f16 fragment (ISA 7.12.2): row m = lane%16, half h = lane/16
// lanes 0-15: K 0..7 (dwords 0-3), K 16..23 (dwords 8-11); lanes 16-31: +8 / +8
__device__ __forceinline__ v16h load_fragA(const unsigned* lds, int lane) {
  int m = lane & 15, h = (lane >> 4) & 1;
  const uint4v* p = (const uint4v*)(lds + m * TROW);
  FragU f;
  f.u4[0] = p[h];
  f.u4[1] = p[2 + h];
  return f.h;
}

// B-matrix 32x16 f16 fragment: column n = lane%16; same K order as A fragment
// staged K-major (row = n, 32 halves of K) -> dwords [4h .. 4h+3], [8+4h .. 8+4h+3] of row n
__device__ __forceinline__ v16h load_fragB(const unsigned* lds, int lane) {
  int n = lane & 15, h = (lane >> 4) & 1;
  const uint4v* p = (const uint4v*)(lds + n * TROW);
  FragU f;
  f.u4[0] = p[h];
  f.u4[1] = p[2 + h];
  return f.h;
}

#define WMMA_F32_F16(a, b, c) \
  __builtin_amdgcn_wmma_f32_16x16x32_f16(false, (a), false, (b), (short)0, (c), false, false)

// CDNA5 async global->LDS copy, 16B per lane, tracked by ASYNCcnt.
#define ASYNC_B128(loff, gptr) \
  asm volatile("global_load_async_to_lds_b128 %0, %1, off" :: "v"(loff), "v"(gptr) : "memory")

__device__ __forceinline__ unsigned lds_off(const void* p) {
  return (unsigned)(uintptr_t)p;   // low 32 bits of generic addr = LDS offset (ISA 10.2)
}

// ---------------- 1) mask compaction (wave per batch) ----------------
__global__ __launch_bounds__(32) void gather_kernel(
    const float* __restrict__ xv, const uint8_t* __restrict__ mask,
    float* __restrict__ g, int* __restrict__ lens)
{
  int b = blockIdx.x, lane = threadIdx.x;
  int cnt = 0;
  for (int base = 0; base < TT; base += 32) {
    int idx = base + lane;
    bool vis = mask[(size_t)b * TT + idx] == 0;
    unsigned bal = (unsigned)__ballot(vis);
    int pre = __popc(bal & ((1u << lane) - 1u));
    if (vis) {
      int pos = cnt + pre;
      if (pos < LVIS) g[(size_t)b * LVIS + pos] = xv[(size_t)b * TT + idx];
    }
    cnt += __popc(bal);
  }
  if (cnt > LVIS) cnt = LVIS;
  if (lane == 0) lens[b] = cnt;
  for (int p = cnt + lane; p < LVIS; p += 32) g[(size_t)b * LVIS + p] = 0.f;
}

// ---------------- 2) input projection + positional encoding ----------------
__global__ void embed_kernel(const float* __restrict__ g, const float* __restrict__ W,
                             const float* __restrict__ Bi, float* __restrict__ X,
                             _Float16* __restrict__ Xh)
{
  int idx = blockIdx.x * blockDim.x + threadIdx.x;
  if (idx >= BB * LVIS * DD) return;
  int d = idx % DD;
  int t = (idx / DD) % LVIS;
  int b = idx / (DD * LVIS);
  int ieven = d & ~1;
  float ang = (float)t * __expf(-(float)ieven * (9.210340371976184f / (float)DD));
  float pe = (d & 1) ? cosf(ang) : sinf(ang);
  float y = g[(size_t)b * LVIS + t] * W[d] + Bi[d] + pe;
  X[idx] = y;
  Xh[idx] = (_Float16)y;
}

// ---------------- 3) FiLM conditioning, all layers ----------------
__global__ void film_kernel(const float* __restrict__ SA, const float* __restrict__ FW,
                            const float* __restrict__ FB, float* __restrict__ GB)
{
  int idx = blockIdx.x * blockDim.x + threadIdx.x;
  if (idx >= NLL * BB * 2 * DD) return;
  int j = idx % (2 * DD);
  int b = (idx / (2 * DD)) % BB;
  int l = idx / (2 * DD * BB);
  float acc = FB[(size_t)l * 2 * DD + j];
  for (int s = 0; s < SDD; ++s)
    acc += SA[b * SDD + s] * FW[((size_t)l * SDD + s) * 2 * DD + j];
  GB[idx] = tanhf(acc);
}

// ---------------- 3b) weight convert + transpose: W(KxN f32) -> Wt(NxK f16) --
__global__ void wconv_kernel(const float* __restrict__ W, _Float16* __restrict__ Wt,
                             int K, int N)
{
  int idx = blockIdx.x * blockDim.x + threadIdx.x;
  if (idx >= K * N) return;
  int k = idx % K, n = idx / K;
  Wt[(size_t)n * K + k] = (_Float16)W[(size_t)k * N + n];
}

// ---------------- 4) WMMA GEMM: C = A(MxK f16) * Wt(NxK f16)^T + bias ---------
// 256 thr = 8 waves; block tile 64x128; wave tile 32x32 (4 WMMA per K-step);
// async b128 global->LDS staging, double-buffered, ASYNCcnt-pipelined.
__global__ __launch_bounds__(256) void gemm_kernel(
    const _Float16* __restrict__ A, const _Float16* __restrict__ Wt,
    const float* __restrict__ bias, float* __restrict__ outF,
    _Float16* __restrict__ outH, int M, int N, int K, int relu)
{
  __shared__ unsigned sA[2][64 * TROW];    // 64 rows x 32 halves
  __shared__ unsigned sB[2][128 * TROW];   // 128 rows x 32 halves (N-major, K runs)
  int tid = threadIdx.x, lane = tid & 31, wave = tid >> 5;
  int bm = blockIdx.y * 64, bn = blockIdx.x * 128;
  int wm = (wave >> 2) * 32;          // 0,32
  int wn = (wave & 3) * 32;           // 0,32,64,96
  // staging assignment: 16B chunks
  int rA = tid >> 2, sgA = tid & 3;           // A: 64 rows x 4 chunks = 1/thread
  int rB = tid >> 2, sgB = tid & 3;           // B: 128 rows x 4 chunks = 2/thread
  const char* baseA  = (const char*)(A  + (size_t)(bm + rA) * K) + sgA * 16;
  const char* baseB0 = (const char*)(Wt + (size_t)(bn + rB) * K) + sgB * 16;
  const char* baseB1 = (const char*)(Wt + (size_t)(bn + 64 + rB) * K) + sgB * 16;
  unsigned offA[2], offB0[2], offB1[2];
  for (int st = 0; st < 2; ++st) {
    offA[st]  = lds_off(&sA[st][0]) + rA * 64 + sgA * 16;
    offB0[st] = lds_off(&sB[st][0]) + rB * 64 + sgB * 16;
    offB1[st] = lds_off(&sB[st][0]) + (64 + rB) * 64 + sgB * 16;
  }
  v8f c00 = {}, c01 = {}, c10 = {}, c11 = {};
  int nk = K >> 5;
  // prologue: stage 0
  ASYNC_B128(offA[0],  baseA);
  ASYNC_B128(offB0[0], baseB0);
  ASYNC_B128(offB1[0], baseB1);
  for (int kt = 0; kt < nk; ++kt) {
    int st = kt & 1;
    if (kt + 1 < nk) {
      size_t go = (size_t)(kt + 1) * 64;     // 32 halves = 64 bytes per K-step
      int sn = st ^ 1;
      ASYNC_B128(offA[sn],  baseA + go);
      ASYNC_B128(offB0[sn], baseB0 + go);
      ASYNC_B128(offB1[sn], baseB1 + go);
      asm volatile("s_wait_asynccnt 3" ::: "memory");
    } else {
      asm volatile("s_wait_asynccnt 0" ::: "memory");
    }
    __syncthreads();
    const unsigned* pA = sA[st];
    const unsigned* pB = sB[st];
    v16h a0 = load_fragA(pA + wm * TROW, lane);
    v16h a1 = load_fragA(pA + (wm + 16) * TROW, lane);
    v16h b0 = load_fragB(pB + wn * TROW, lane);
    v16h b1 = load_fragB(pB + (wn + 16) * TROW, lane);
    c00 = WMMA_F32_F16(a0, b0, c00);
    c01 = WMMA_F32_F16(a0, b1, c01);
    c10 = WMMA_F32_F16(a1, b0, c10);
    c11 = WMMA_F32_F16(a1, b1, c11);
    __syncthreads();
  }
  int hh = lane >> 4, nlo = lane & 15;
  for (int i = 0; i < 8; ++i) {
    int r0 = bm + wm + i + 8 * hh;
    int r1 = r0 + 16;
    int c0 = bn + wn + nlo, c1 = c0 + 16;
    float v00 = c00[i] + bias[c0], v01 = c01[i] + bias[c1];
    float v10 = c10[i] + bias[c0], v11 = c11[i] + bias[c1];
    if (relu) {
      v00 = fmaxf(v00, 0.f); v01 = fmaxf(v01, 0.f);
      v10 = fmaxf(v10, 0.f); v11 = fmaxf(v11, 0.f);
    }
    if (outF) {
      outF[(size_t)r0 * N + c0] = v00; outF[(size_t)r0 * N + c1] = v01;
      outF[(size_t)r1 * N + c0] = v10; outF[(size_t)r1 * N + c1] = v11;
    }
    if (outH) {
      outH[(size_t)r0 * N + c0] = (_Float16)v00; outH[(size_t)r0 * N + c1] = (_Float16)v01;
      outH[(size_t)r1 * N + c0] = (_Float16)v10; outH[(size_t)r1 * N + c1] = (_Float16)v11;
    }
  }
}

// ---------------- 5) flash attention (f16 in/out) -----------------------------
// block = 8 waves, covers (b, h, 128 q rows); K/V streamed in 32-token chunks.
__global__ __launch_bounds__(256) void attn_kernel(
    const _Float16* __restrict__ Qh, const _Float16* __restrict__ Kh,
    const _Float16* __restrict__ Vh, const int* __restrict__ lens,
    _Float16* __restrict__ O)
{
  int bid = blockIdx.x;
  int qt = bid & 3;
  int h  = (bid >> 2) & 7;
  int b  = bid >> 5;
  int tid = threadIdx.x, lane = tid & 31, wave = tid >> 5;
  int len = lens[b];
  __shared__ unsigned sQ[128 * TROW];      // 128 q-rows x 32 dims
  __shared__ unsigned sK[32 * TROW];       // 32 tokens x 32 dims  (B-frag rows = token)
  __shared__ unsigned sV[32 * TROW];       // 32 dims  x 32 tokens (B-frag rows = dim)
  __shared__ unsigned sP[8][16 * TROW];    // per-wave P tile 16x32
  const int dh = h * DHH;
  int q0 = qt * 128;
  // ---- stage Q tile via async b128 (2 chunks / thread) ----
  unsigned qbase = lds_off(&sQ[0]);
  for (int i = 0; i < 2; ++i) {
    int ch = tid + 256 * i;
    int r = ch >> 2, s = ch & 3;
    const char* g = (const char*)(Qh + ((size_t)b * LVIS + q0 + r) * DD + dh) + s * 16;
    ASYNC_B128(qbase + r * 64 + s * 16, g);
  }
  asm volatile("s_wait_asynccnt 0" ::: "memory");
  __syncthreads();
  v16h qf = load_fragA(sQ + (wave * 16) * TROW, lane);
  v8f o0 = {}, o1 = {};
  float mrow[8], lrow[8];
  for (int i = 0; i < 8; ++i) { mrow[i] = -1e30f; lrow[i] = 0.f; }
  const float scale = 0.17677669529663687f;   // 1/sqrt(32)
  _Float16* hV = (_Float16*)sV;
  unsigned kbase = lds_off(&sK[0]);
  int hh = lane >> 4, nlo = lane & 15;
  int dg = tid & 7, tt = tid >> 3;            // V staging: 4 dims x 1 token per thread
  for (int kc = 0; kc < LVIS; kc += 32) {
    if (tid < 128) {                          // K chunk: 32 rows x 4 chunks, async
      int r = tid >> 2, s = tid & 3;
      const char* g = (const char*)(Kh + ((size_t)b * LVIS + kc + r) * DD + dh) + s * 16;
      ASYNC_B128(kbase + tid * 16, g);
    }
    {                                          // V chunk: transposed manual stage
      union { uint2v u; _Float16 hx[4]; } vv;
      vv.u = *(const uint2v*)(Vh + ((size_t)b * LVIS + kc + tt) * DD + dh + dg * 4);
      for (int j = 0; j < 4; ++j) hV[(dg * 4 + j) * 32 + tt] = vv.hx[j];
    }
    asm volatile("s_wait_asynccnt 0" ::: "memory");
    __syncthreads();
    v16h kf0 = load_fragB(sK, lane);
    v16h kf1 = load_fragB(sK + 16 * TROW, lane);
    v8f z = {};
    v8f s0 = WMMA_F32_F16(qf, kf0, z);
    v8f s1 = WMMA_F32_F16(qf, kf1, z);
    int tok0 = kc + nlo, tok1 = tok0 + 16;
    bool m0 = tok0 >= len, m1 = tok1 >= len;
    _Float16* myP = (_Float16*)sP[wave];
    for (int i = 0; i < 8; ++i) {
      float a0 = m0 ? -1e9f : s0[i] * scale;
      float a1 = m1 ? -1e9f : s1[i] * scale;
      float v = fmaxf(a0, a1);                 // row max across the 16-lane half
      for (int off = 8; off; off >>= 1) v = fmaxf(v, __shfl_xor(v, off, 16));
      float mn = fmaxf(mrow[i], v);
      float f  = __expf(mrow[i] - mn);
      float p0 = __expf(a0 - mn);
      float p1 = __expf(a1 - mn);
      float rs = p0 + p1;
      for (int off = 8; off; off >>= 1) rs += __shfl_xor(rs, off, 16);
      lrow[i] = lrow[i] * f + rs;
      mrow[i] = mn;
      o0[i] *= f; o1[i] *= f;
      int pr = i + 8 * hh;                     // C-layout row -> P tile row
      myP[pr * 32 + nlo]      = (_Float16)p0;
      myP[pr * 32 + nlo + 16] = (_Float16)p1;
    }
    v16h pf  = load_fragA(sP[wave], lane);     // same-wave LDS ops are in-order
    v16h vf0 = load_fragB(sV, lane);
    v16h vf1 = load_fragB(sV + 16 * TROW, lane);
    o0 = WMMA_F32_F16(pf, vf0, o0);
    o1 = WMMA_F32_F16(pf, vf1, o1);
    __syncthreads();
  }
  for (int i = 0; i < 8; ++i) {
    float inv = 1.f / lrow[i];
    int qrow = q0 + wave * 16 + i + 8 * hh;
    size_t base = ((size_t)b * LVIS + qrow) * DD + dh;
    O[base + nlo]      = (_Float16)(o0[i] * inv);
    O[base + nlo + 16] = (_Float16)(o1[i] * inv);
  }
}

// ---------------- 6) fused residual + LayerNorm (+ FiLM), f32 + f16 out ------
__global__ __launch_bounds__(256) void ln_kernel(
    const float* __restrict__ X, const float* __restrict__ Rsd,
    const float* __restrict__ G, const float* __restrict__ Bb,
    const float* __restrict__ film, float* __restrict__ Out,
    _Float16* __restrict__ OutH, int out_batch_stride)
{
  int wave = threadIdx.x >> 5, lane = threadIdx.x & 31;
  int row = blockIdx.x * 8 + wave;
  int b = row / LVIS, t = row % LVIS;
  size_t base = (size_t)row * DD;
  float v[8], s = 0.f, s2 = 0.f;
  for (int i = 0; i < 8; ++i) {
    int d = i * 32 + lane;
    float x = X[base + d];
    if (Rsd) x += Rsd[base + d];
    v[i] = x; s += x; s2 += x * x;
  }
  for (int off = 16; off; off >>= 1) {
    s  += __shfl_xor(s,  off, 32);
    s2 += __shfl_xor(s2, off, 32);
  }
  float mu  = s * (1.f / DD);
  float var = s2 * (1.f / DD) - mu * mu;
  float r = rsqrtf(var + 1e-5f);
  size_t ob = (size_t)b * out_batch_stride + (size_t)t * DD;
  for (int i = 0; i < 8; ++i) {
    int d = i * 32 + lane;
    float y = (v[i] - mu) * r * G[d] + Bb[d];
    if (film) y = film[(size_t)b * 2 * DD + d] * y + film[(size_t)b * 2 * DD + DD + d];
    Out[ob + d] = y;
    if (OutH) OutH[base + d] = (_Float16)y;
  }
}

// ---------------- 7) static token row ----------------
__global__ void static_token_kernel(const float* __restrict__ SA, const float* __restrict__ AW,
                                    const float* __restrict__ AB, float* __restrict__ Out)
{
  int idx = blockIdx.x * blockDim.x + threadIdx.x;
  if (idx >= BB * DD) return;
  int d = idx % DD, b = idx / DD;
  float acc = AB[d];
  for (int s = 0; s < SDD; ++s) acc += SA[b * SDD + s] * AW[s * DD + d];
  Out[((size_t)b * (LVIS + 1) + LVIS) * DD + d] = acc;
}

// ---------------- host side ----------------
extern "C" void kernel_launch(void* const* d_in, const int* in_sizes, int n_in,
                              void* d_out, int out_size, void* d_ws, size_t ws_size,
                              hipStream_t stream) {
  const float*   x_vec  = (const float*)d_in[0];
  const float*   sattr  = (const float*)d_in[1];
  const uint8_t* tmask  = (const uint8_t*)d_in[2];
  const float* in_w   = (const float*)d_in[3];
  const float* in_b   = (const float*)d_in[4];
  const float* film_w = (const float*)d_in[5];
  const float* film_b = (const float*)d_in[6];
  const float* wq = (const float*)d_in[7];   const float* bq = (const float*)d_in[8];
  const float* wk = (const float*)d_in[9];   const float* bk = (const float*)d_in[10];
  const float* wv = (const float*)d_in[11];  const float* bv = (const float*)d_in[12];
  const float* wo = (const float*)d_in[13];  const float* bo = (const float*)d_in[14];
  const float* ln1g = (const float*)d_in[15]; const float* ln1b = (const float*)d_in[16];
  const float* w1 = (const float*)d_in[17];  const float* b1 = (const float*)d_in[18];
  const float* w2 = (const float*)d_in[19];  const float* b2 = (const float*)d_in[20];
  const float* ln2g = (const float*)d_in[21]; const float* ln2b = (const float*)d_in[22];
  const float* ng = (const float*)d_in[23];  const float* nb = (const float*)d_in[24];
  const float* aw = (const float*)d_in[25];  const float* ab = (const float*)d_in[26];
  float* out = (float*)d_out;

  const size_t MBy = 1ull << 20;
  char* ws = (char*)d_ws;
  float*    x    = (float*)   (ws + 0 * MBy);   // 16 MB master f32 activations
  _Float16* xh   = (_Float16*)(ws + 16 * MBy);  // 8 MB f16 copy
  _Float16* qh   = (_Float16*)(ws + 24 * MBy);  // 8 MB
  _Float16* kh   = (_Float16*)(ws + 32 * MBy);  // 8 MB
  _Float16* vh   = (_Float16*)(ws + 40 * MBy);  // 8 MB
  _Float16* th   = (_Float16*)(ws + 48 * MBy);  // 8 MB attention out
  float*    rb   = (float*)   (ws + 56 * MBy);  // 16 MB residual (O-proj / FFN2 out)
  _Float16* ffh  = (_Float16*)(ws + 24 * MBy);  // 32 MB FFN hidden (aliases qh..th, dead)
  _Float16* wqT  = (_Float16*)(ws + 72 * MBy);            // NL x 256x256
  _Float16* wkT  = (_Float16*)(ws + 72 * MBy + 512 * 1024);
  _Float16* wvT  = (_Float16*)(ws + 73 * MBy);
  _Float16* woT  = (_Float16*)(ws + 73 * MBy + 512 * 1024);
  _Float16* w1T  = (_Float16*)(ws + 74 * MBy);            // NL x 1024x256
  _Float16* w2T  = (_Float16*)(ws + 76 * MBy);            // NL x 256x1024
  float*    gbuf = (float*)   (ws + 78 * MBy);            // NL*B*2D
  int*      lens = (int*)     (ws + 78 * MBy + 256 * 1024);
  float*    gvec = (float*)   (ws + 78 * MBy + 512 * 1024);

  const int M = BB * LVIS;   // 16384 activation rows

  gather_kernel<<<BB, 32, 0, stream>>>(x_vec, tmask, gvec, lens);
  embed_kernel<<<(BB * LVIS * DD + 255) / 256, 256, 0, stream>>>(gvec, in_w, in_b, x, xh);
  film_kernel<<<(NLL * BB * 2 * DD + 255) / 256, 256, 0, stream>>>(sattr, film_w, film_b, gbuf);

  // weight convert+transpose (f32 KxN -> f16 NxK)
  for (int l = 0; l < NLL; ++l) {
    int gDD = (DD * DD + 255) / 256, gFF = (DD * DFFF + 255) / 256;
    wconv_kernel<<<gDD, 256, 0, stream>>>(wq + (size_t)l * DD * DD,  wqT + (size_t)l * DD * DD,  DD, DD);
    wconv_kernel<<<gDD, 256, 0, stream>>>(wk + (size_t)l * DD * DD,  wkT + (size_t)l * DD * DD,  DD, DD);
    wconv_kernel<<<gDD, 256, 0, stream>>>(wv + (size_t)l * DD * DD,  wvT + (size_t)l * DD * DD,  DD, DD);
    wconv_kernel<<<gDD, 256, 0, stream>>>(wo + (size_t)l * DD * DD,  woT + (size_t)l * DD * DD,  DD, DD);
    wconv_kernel<<<gFF, 256, 0, stream>>>(w1 + (size_t)l * DD * DFFF, w1T + (size_t)l * DD * DFFF, DD, DFFF);
    wconv_kernel<<<gFF, 256, 0, stream>>>(w2 + (size_t)l * DFFF * DD, w2T + (size_t)l * DFFF * DD, DFFF, DD);
  }

  for (int l = 0; l < NLL; ++l) {
    dim3 gProj(DD / 128, M / 64);
    gemm_kernel<<<gProj, 256, 0, stream>>>(xh, wqT + (size_t)l * DD * DD, bq + l * DD,
                                           nullptr, qh, M, DD, DD, 0);
    gemm_kernel<<<gProj, 256, 0, stream>>>(xh, wkT + (size_t)l * DD * DD, bk + l * DD,
                                           nullptr, kh, M, DD, DD, 0);
    gemm_kernel<<<gProj, 256, 0, stream>>>(xh, wvT + (size_t)l * DD * DD, bv + l * DD,
                                           nullptr, vh, M, DD, DD, 0);

    attn_kernel<<<BB * NHH * (LVIS / 128), 256, 0, stream>>>(qh, kh, vh, lens, th);

    gemm_kernel<<<gProj, 256, 0, stream>>>(th, woT + (size_t)l * DD * DD, bo + l * DD,
                                           rb, nullptr, M, DD, DD, 0);

    ln_kernel<<<M / 8, 256, 0, stream>>>(x, rb, ln1g + l * DD, ln1b + l * DD,
                                         gbuf + (size_t)l * BB * 2 * DD, x, xh, LVIS * DD);

    dim3 gF1(DFFF / 128, M / 64);
    gemm_kernel<<<gF1, 256, 0, stream>>>(xh, w1T + (size_t)l * DD * DFFF, b1 + l * DFFF,
                                         nullptr, ffh, M, DFFF, DD, 1);
    dim3 gF2(DD / 128, M / 64);
    gemm_kernel<<<gF2, 256, 0, stream>>>(ffh, w2T + (size_t)l * DFFF * DD, b2 + l * DD,
                                         rb, nullptr, M, DD, DFFF, 0);

    ln_kernel<<<M / 8, 256, 0, stream>>>(x, rb, ln2g + l * DD, ln2b + l * DD,
                                         nullptr, x, xh, LVIS * DD);
  }

  ln_kernel<<<M / 8, 256, 0, stream>>>(x, nullptr, ng, nb, nullptr, out, nullptr,
                                       (LVIS + 1) * DD);
  static_token_kernel<<<(BB * DD + 255) / 256, 256, 0, stream>>>(sattr, aw, ab, out);
}
